// StructureExtractor_71184787964506
// MI455X (gfx1250) — matrix operand order repaired and checked by reading, not compiled, not run on hardware
//
#include <hip/hip_runtime.h>

typedef __attribute__((ext_vector_type(16))) _Float16 v16h;
typedef __attribute__((ext_vector_type(8)))  _Float16 v8h;
typedef __attribute__((ext_vector_type(8)))  float    v8f;

#define HH     480
#define WW     640
#define SCALE8 8
#define LCOLS  80
#define LVAL   4800
#define NB     4
#define CC     256
#define AA     128
#define PP     (HH*WW)
#define DD     768
#define MTOT   (2*NB*LVAL)   /* 38400 rows: [aug0 rows | aug1 rows] */

// ---------------- weight transpose + f32->f16 convert ----------------
// dst[n*K + k] = (f16) src[k*N + n]   (src is [K][N] row-major)
__global__ void convert_transpose_kernel(const float* __restrict__ src,
                                         _Float16* __restrict__ dst,
                                         int K, int N) {
  int idx = blockIdx.x * blockDim.x + threadIdx.x;
  if (idx >= K * N) return;
  int n = idx / K;
  int k = idx - n * K;
  dst[idx] = (_Float16)src[k * N + n];
}

// ---------------- pack feat0/feat1 into X columns [0,256) ----------------
__global__ void feat_pack_kernel(const float* __restrict__ feat0,
                                 const float* __restrict__ feat1,
                                 _Float16* __restrict__ X) {
  int idx = blockIdx.x * blockDim.x + threadIdx.x;   // exactly MTOT*CC threads
  int r = idx >> 8;          // / 256
  int c = idx & 255;
  float v = (r < MTOT / 2) ? feat0[(size_t)r * CC + c]
                           : feat1[(size_t)(r - MTOT / 2) * CC + c];
  X[(size_t)r * DD + c] = (_Float16)v;
}

// ---------------- struct feats into X columns [256,768) ----------------
__global__ __launch_bounds__(128)
void struct_feats_kernel(const float* __restrict__ pts,  // [NB][PP][3]
                         const int*   __restrict__ ids,  // [NB][AA]
                         _Float16* __restrict__ X,       // [MTOT][DD]
                         int rowBase) {
  __shared__ float ax[AA], ay[AA], az[AA];
  const int n   = blockIdx.y;
  const int tid = threadIdx.x;
  if (tid < AA) {
    int id = ids[n * AA + tid];
    const float* p = pts + ((size_t)n * PP + (size_t)id) * 3;
    ax[tid] = p[0]; ay[tid] = p[1]; az[tid] = p[2];
  }
  __syncthreads();
  int l = blockIdx.x * blockDim.x + tid;
  if (l >= LVAL) return;
  int lr = l / LCOLS, lc = l - lr * LCOLS;
  size_t pidx = ((size_t)n * PP + (size_t)(lr * SCALE8) * WW + (size_t)(lc * SCALE8)) * 3;
  float px = pts[pidx], py = pts[pidx + 1], pz = pts[pidx + 2];
  float s0 = 0.f, s1 = 0.f, s2 = 0.f, s3 = 0.f;
  for (int a = 0; a < AA; ++a) {
    float dx = px - ax[a], dy = py - ay[a], dz = pz - az[a];
    float d2 = dx * dx + dy * dy + dz * dz;
    s0 += fabsf(dx); s1 += fabsf(dy); s2 += fabsf(dz); s3 += d2;
  }
  float i0 = 1.0f / s0, i1 = 1.0f / s1, i2 = 1.0f / s2, i3 = 1.0f / s3;
  _Float16* xr = X + (size_t)(rowBase + n * LVAL + l) * DD + CC;
  for (int a = 0; a < AA; ++a) {
    float dx = px - ax[a], dy = py - ay[a], dz = pz - az[a];
    float d2 = dx * dx + dy * dy + dz * dz;
    xr[0 * AA + a] = (_Float16)(dx * i0);
    xr[1 * AA + a] = (_Float16)(dy * i1);
    xr[2 * AA + a] = (_Float16)(dz * i2);
    xr[3 * AA + a] = (_Float16)(d2 * i3);
  }
}

// ---------------- tiled WMMA GEMM: out = act(A @ B + bias) ----------------
// A  : [M][K]  f16 row-major
// Bt : [N][K]  f16 row-major (i.e. B transposed; B is [K][N])
// Double-buffered LDS tiles filled with GLOBAL_LOAD_ASYNC_TO_LDS_B128
// (ASYNCcnt-tracked), one workgroup barrier per K-step.
template <bool RELU16>
__global__ __launch_bounds__(256)
void gemm_wmma_kernel(const _Float16* __restrict__ A,
                      const _Float16* __restrict__ Bt,
                      const float*    __restrict__ bias,
                      _Float16* __restrict__ outH,
                      float*    __restrict__ outF,
                      int K, int Ncols) {
  __shared__ _Float16 As[2][128][32];
  __shared__ _Float16 Bs[2][128][32];

  const int tid   = threadIdx.x;
  const int lane  = tid & 31;
  const int wave  = tid >> 5;
  const int waveM = wave & 3;   // 4 waves along M -> 32 rows each
  const int waveN = wave >> 2;  // 2 waves along N -> 64 cols each
  const int m0 = blockIdx.x * 128;
  const int n0 = blockIdx.y * 128;

  v8f acc[2][4] = {};

  // async-copy slots: thread covers (lrow, lhoff..lhoff+16) = 2x16B per matrix
  const int lrow  = tid >> 1;         // 0..127
  const int lhoff = (tid & 1) * 16;   // 0 or 16 halves
  const _Float16* ga = A  + (size_t)(m0 + lrow) * K + lhoff;
  const _Float16* gb = Bt + (size_t)(n0 + lrow) * K + lhoff;

  // LDS byte addresses (flat-pointer low 32 bits == LDS offset per ISA aperture rule)
  uint32_t ldsA[2], ldsB[2];
  ldsA[0] = (uint32_t)(uintptr_t)&As[0][lrow][lhoff];
  ldsA[1] = (uint32_t)(uintptr_t)&As[1][lrow][lhoff];
  ldsB[0] = (uint32_t)(uintptr_t)&Bs[0][lrow][lhoff];
  ldsB[1] = (uint32_t)(uintptr_t)&Bs[1][lrow][lhoff];

  // per-lane fragment addressing (CDNA5 wave32 WMMA layouts)
  const int aRow  = waveM * 32 + (lane & 15);
  const int koffA = (lane >> 4) * 8;     // A: halves [koffA..+8) and [16+koffA..+8)
  const int bRow  = waveN * 64 + (lane & 15);
  const int kbB   = (lane >> 4) * 8;     // B: halves [kbB..+8) and [16+kbB..+8)

  // prologue: async-copy K-tile 0 into buffer 0
  asm volatile(
      "global_load_async_to_lds_b128 %0, %2, off\n\t"
      "global_load_async_to_lds_b128 %0, %2, off offset:16\n\t"
      "global_load_async_to_lds_b128 %1, %3, off\n\t"
      "global_load_async_to_lds_b128 %1, %3, off offset:16"
      :: "v"(ldsA[0]), "v"(ldsB[0]), "v"(ga), "v"(gb) : "memory");
  asm volatile("s_wait_asynccnt 0x0" ::: "memory");
  __syncthreads();

  int buf = 0;
  for (int k0 = 0; k0 < K; k0 += 32) {
    const int  nb   = buf ^ 1;
    const bool more = (k0 + 32) < K;
    if (more) {   // kick off next tile into the other buffer (overlaps with WMMAs)
      const _Float16* gan = ga + k0 + 32;
      const _Float16* gbn = gb + k0 + 32;
      asm volatile(
          "global_load_async_to_lds_b128 %0, %2, off\n\t"
          "global_load_async_to_lds_b128 %0, %2, off offset:16\n\t"
          "global_load_async_to_lds_b128 %1, %3, off\n\t"
          "global_load_async_to_lds_b128 %1, %3, off offset:16"
          :: "v"(ldsA[nb]), "v"(ldsB[nb]), "v"(gan), "v"(gbn) : "memory");
    }

    union Frag { v16h v; v8h h[2]; };
    Frag af[2], bf[4];
#pragma unroll
    for (int mi = 0; mi < 2; ++mi) {
      af[mi].h[0] = *(const v8h*)&As[buf][aRow + mi * 16][koffA];
      af[mi].h[1] = *(const v8h*)&As[buf][aRow + mi * 16][16 + koffA];
    }
#pragma unroll
    for (int ni = 0; ni < 4; ++ni) {
      bf[ni].h[0] = *(const v8h*)&Bs[buf][bRow + ni * 16][kbB];
      bf[ni].h[1] = *(const v8h*)&Bs[buf][bRow + ni * 16][kbB + 16];
    }
#pragma unroll
    for (int mi = 0; mi < 2; ++mi)
#pragma unroll
      for (int ni = 0; ni < 4; ++ni)
        acc[mi][ni] = __builtin_amdgcn_wmma_f32_16x16x32_f16(
            false, af[mi].v, false, bf[ni].v, (short)0, acc[mi][ni], false, false);

    if (more) asm volatile("s_wait_asynccnt 0x0" ::: "memory");
    __syncthreads();   // next buffer filled everywhere, current buffer fully read
    buf = nb;
  }

  // epilogue: C/D layout -> element r of v8f is row (lane>=16 ? 8+r : r), col lane%16
  const int mBase = m0 + waveM * 32 + (lane >> 4) * 8;
  const int nBase = n0 + waveN * 64 + (lane & 15);
#pragma unroll
  for (int mi = 0; mi < 2; ++mi)
#pragma unroll
    for (int ni = 0; ni < 4; ++ni) {
      int col = nBase + ni * 16;
      float bv = bias[col];
#pragma unroll
      for (int r = 0; r < 8; ++r) {
        int row = mBase + mi * 16 + r;
        float v = acc[mi][ni][r] + bv;
        if (RELU16)
          outH[(size_t)row * Ncols + col] = (_Float16)(v > 0.f ? v : 0.f);
        else
          outF[(size_t)row * Ncols + col] = v;
      }
    }
}

extern "C" void kernel_launch(void* const* d_in, const int* in_sizes, int n_in,
                              void* d_out, int out_size, void* d_ws, size_t ws_size,
                              hipStream_t stream) {
  const float* feat0 = (const float*)d_in[0];
  const float* feat1 = (const float*)d_in[1];
  const float* pts0  = (const float*)d_in[2];
  const float* pts1  = (const float*)d_in[3];
  const int*   ids0  = (const int*)d_in[4];
  const int*   ids1  = (const int*)d_in[5];
  const float* W1    = (const float*)d_in[6];
  const float* b1    = (const float*)d_in[7];
  const float* W2    = (const float*)d_in[8];
  const float* b2    = (const float*)d_in[9];
  float* out = (float*)d_out;

  // workspace carve-up (f16): W1t | W2t | X | H  (~120 MB)
  _Float16* ws  = (_Float16*)d_ws;
  _Float16* W1t = ws;                    ws += (size_t)DD * DD;
  _Float16* W2t = ws;                    ws += (size_t)CC * DD;
  _Float16* X   = ws;                    ws += (size_t)MTOT * DD;
  _Float16* Hh  = ws;                    ws += (size_t)MTOT * DD;

  convert_transpose_kernel<<<(DD * DD + 255) / 256, 256, 0, stream>>>(W1, W1t, DD, DD);
  convert_transpose_kernel<<<(DD * CC + 255) / 256, 256, 0, stream>>>(W2, W2t, DD, CC);
  feat_pack_kernel<<<(MTOT * CC) / 256, 256, 0, stream>>>(feat0, feat1, X);

  dim3 sg((LVAL + 127) / 128, NB);
  struct_feats_kernel<<<sg, 128, 0, stream>>>(pts0, ids0, X, 0);
  struct_feats_kernel<<<sg, 128, 0, stream>>>(pts1, ids1, X, MTOT / 2);

  dim3 g1(MTOT / 128, DD / 128);   // 300 x 6
  gemm_wmma_kernel<true><<<g1, 256, 0, stream>>>(X, W1t, b1, Hh, nullptr, DD, DD);

  dim3 g2(MTOT / 128, CC / 128);   // 300 x 2
  gemm_wmma_kernel<false><<<g2, 256, 0, stream>>>(Hh, W2t, b2, nullptr, out, DD, CC);
}
